// GNNArm_54417235640479
// MI455X (gfx1250) — matrix operand order repaired and checked by reading, not compiled, not run on hardware
//
#include <hip/hip_runtime.h>
#include <hip/hip_bf16.h>

typedef __bf16 bf16_t;
typedef __attribute__((ext_vector_type(16))) __bf16 v16bf;
typedef __attribute__((ext_vector_type(8)))  float  v8f;

#define N_NODES  50000
#define N_EDGES  800000
#define E_TOT    (N_EDGES + N_NODES)   // edges + self loops = 850000
#define N_GRAPHS 512
#define IN_CH    58
#define D1       64
#define HEADS1   4
#define HD1      256                   // HEADS1*D1
#define OUT_CH   128
#define MT       (N_NODES / 16)        // 3125 row tiles (50000 % 16 == 0)
#define KB1      2                     // K=64  -> 2 k-blocks of 32
#define KB2      8                     // K=256 -> 8 k-blocks of 32
#define NEG_SLOPE 0.2f
#define BN_EPS   1e-5f

// ---------- order-preserving float<->uint key (for atomic max on floats) ----
__device__ __forceinline__ unsigned fkey(float f) {
    unsigned b = __float_as_uint(f);
    return (b & 0x80000000u) ? ~b : (b | 0x80000000u);
}
__device__ __forceinline__ float fdecode(unsigned u) {
    unsigned b = (u & 0x80000000u) ? (u & 0x7FFFFFFFu) : ~u;
    return __uint_as_float(b);
}
__device__ __forceinline__ float lrelu(float v) { return v >= 0.f ? v : NEG_SLOPE * v; }

// ---------- pack x [N,58] f32 -> bf16 A-fragment order, K padded to 64 -----
// A-fragment (16-bit 16x32): lane = half*16 + (m&15); within a lane's 16
// elements j: k = (j>>3)*16 + half*8 + (j&7).
__global__ void k_pack_x(const float* __restrict__ x, bf16_t* __restrict__ xp) {
    int tid = blockIdx.x * blockDim.x + threadIdx.x;
    if (tid >= N_NODES * 64) return;
    int n = tid >> 6, k = tid & 63;
    float v = (k < IN_CH) ? x[n * IN_CH + k] : 0.f;
    int mt = n >> 4, lm = n & 15;
    int kb = k >> 5, kloc = k & 31;
    int lane = ((kloc >> 3) & 1) * 16 + lm;
    int j = (kloc >> 4) * 8 + (kloc & 7);
    xp[((size_t)(mt * KB1 + kb) * 32 + lane) * 16 + j] = (bf16_t)v;
}

// ---------- pack W [Ksrc,Ncols] f32 -> bf16 B-fragment order ---------------
// B-fragment (16-bit 32x16): lane = ((kloc>>3)&1)*16 + (n&15); element j: k = (j>>3)*16 + half*8 + (j&7).
__global__ void k_pack_w(const float* __restrict__ W, bf16_t* __restrict__ wp,
                         int Ksrc, int Kpad, int Ncols, int KB) {
    int tid = blockIdx.x * blockDim.x + threadIdx.x;
    if (tid >= Kpad * Ncols) return;
    int k = tid / Ncols, n = tid % Ncols;
    float v = (k < Ksrc) ? W[k * Ncols + n] : 0.f;
    int tile = n >> 4, ln = n & 15;
    int kb = k >> 5, krem = k & 31;
    int lane = ((krem >> 3) & 1) * 16 + ln;
    int j = (krem >> 4) * 8 + (krem & 7);
    wp[((size_t)(tile * KB + kb) * 32 + lane) * 16 + j] = (bf16_t)v;
}

// ---------- WMMA GEMM: two 16x16 column tiles per wave, shared A fragment --
// Two independent accumulator chains -> consecutive v_wmma ops have no
// D->C RAW dependency and can fill the XDL pipe back-to-back (ISA 7.12.1).
__global__ void k_gemm(const bf16_t* __restrict__ Ap, const bf16_t* __restrict__ Bp,
                       float* __restrict__ C, int KB, int ldc) {
    int wave = threadIdx.x >> 5;
    int lane = threadIdx.x & 31;
    int mt = blockIdx.x;
    int ct0 = (blockIdx.y * (blockDim.x >> 5) + wave) * 2;
    const v16bf* ap  = (const v16bf*)Ap + (size_t)(mt * KB) * 32 + lane;
    const v16bf* b0p = (const v16bf*)Bp + (size_t)(ct0 * KB) * 32 + lane;
    const v16bf* b1p = b0p + (size_t)KB * 32;
    v8f c0 = {}, c1 = {};
    for (int kb = 0; kb < KB; ++kb) {
        v16bf a  = ap[kb * 32];
        v16bf b0 = b0p[kb * 32];
        v16bf b1 = b1p[kb * 32];
        c0 = __builtin_amdgcn_wmma_f32_16x16x32_bf16(
                 false, a, false, b0, (short)0, c0, false, false);
        c1 = __builtin_amdgcn_wmma_f32_16x16x32_bf16(
                 false, a, false, b1, (short)0, c1, false, false);
    }
    // C layout: VGPR r -> M = r + (lane>>4)*8 ; N = lane&15
    int row0 = mt * 16 + (lane >> 4) * 8;
    int col  = ct0 * 16 + (lane & 15);
    float* out = C + (size_t)row0 * ldc + col;
#pragma unroll
    for (int r = 0; r < 8; ++r) {
        out[(size_t)r * ldc]      = c0[r];
        out[(size_t)r * ldc + 16] = c1[r];
    }
}

// ---------- attention coefficients a_s/a_d per (node, head) ----------------
__global__ void k_att(const float* __restrict__ h, const float* __restrict__ att_s,
                      const float* __restrict__ att_d, float* __restrict__ a_s,
                      float* __restrict__ a_d, int heads, int D) {
    int tid = blockIdx.x * blockDim.x + threadIdx.x;
    if (tid >= N_NODES * heads) return;
    int n = tid / heads, hh = tid % heads;
    const float* hp = h + (size_t)n * heads * D + hh * D;
    const float* as = att_s + hh * D;
    const float* ad = att_d + hh * D;
    float s0 = 0.f, s1 = 0.f;
    for (int d = 0; d < D; ++d) { float v = hp[d]; s0 += v * as[d]; s1 += v * ad[d]; }
    a_s[tid] = s0;
    a_d[tid] = s1;
}

__device__ __forceinline__ void edge_ends(const int* __restrict__ ei, int e,
                                          int& src, int& dst) {
    if (e < N_EDGES) { src = ei[e]; dst = ei[N_EDGES + e]; }
    else             { src = dst = e - N_EDGES; }
}

// ---------- segment max of leaky-relu(e) by dst ----------------------------
__global__ void k_edge_max(const int* __restrict__ ei, const float* __restrict__ a_s,
                           const float* __restrict__ a_d, unsigned* __restrict__ mkey,
                           int heads) {
    int tid = blockIdx.x * blockDim.x + threadIdx.x;
    if (tid >= E_TOT * heads) return;
    int e = tid / heads, hh = tid % heads;
    int src, dst; edge_ends(ei, e, src, dst);
    float v = lrelu(a_s[src * heads + hh] + a_d[dst * heads + hh]);
    atomicMax(&mkey[dst * heads + hh], fkey(v));
}

// ---------- segment sum of exp(e - max) by dst -----------------------------
__global__ void k_edge_expsum(const int* __restrict__ ei, const float* __restrict__ a_s,
                              const float* __restrict__ a_d,
                              const unsigned* __restrict__ mkey,
                              float* __restrict__ z, int heads) {
    int tid = blockIdx.x * blockDim.x + threadIdx.x;
    if (tid >= E_TOT * heads) return;
    int e = tid / heads, hh = tid % heads;
    int src, dst; edge_ends(ei, e, src, dst);
    float v = lrelu(a_s[src * heads + hh] + a_d[dst * heads + hh]);
    float mf = fdecode(mkey[dst * heads + hh]);
    atomicAdd(&z[dst * heads + hh], __expf(v - mf));
}

// ---------- weighted aggregation: one wave per (edge, head) ----------------
__global__ void k_edge_agg(const int* __restrict__ ei, const float* __restrict__ a_s,
                           const float* __restrict__ a_d,
                           const unsigned* __restrict__ mkey,
                           const float* __restrict__ z,
                           const float* __restrict__ hpre, float* __restrict__ agg,
                           int heads, int D) {
    int gw = (blockIdx.x * blockDim.x + threadIdx.x) >> 5;
    int lane = threadIdx.x & 31;
    if (gw >= E_TOT * heads) return;
    int e = gw / heads, hh = gw % heads;
    int src, dst; edge_ends(ei, e, src, dst);
    int ai = dst * heads + hh;
    float v = lrelu(a_s[src * heads + hh] + a_d[ai]);
    float alpha = __expf(v - fdecode(mkey[ai])) / (z[ai] + 1e-16f);
    const float* hp = hpre + (size_t)src * heads * D + hh * D;
    float* op = agg + (size_t)dst * heads * D + hh * D;
    for (int j = lane; j < D; j += 32)
        atomicAdd(&op[j], hp[j] * alpha);
}

// ---------- layer1 epilog: h1 = relu(agg + b1), repack to bf16 A-fragments -
__global__ void k_bias_relu_pack1(const float* __restrict__ agg,
                                  const float* __restrict__ b,
                                  bf16_t* __restrict__ h1p) {
    int tid = blockIdx.x * blockDim.x + threadIdx.x;
    if (tid >= N_NODES * HD1) return;
    int n = tid >> 8, k = tid & 255;
    float v = agg[tid] + b[k];
    v = v > 0.f ? v : 0.f;
    int mt = n >> 4, lm = n & 15;
    int kb = k >> 5, kloc = k & 31;
    int lane = ((kloc >> 3) & 1) * 16 + lm;
    int j = (kloc >> 4) * 8 + (kloc & 7);
    h1p[((size_t)(mt * KB2 + kb) * 32 + lane) * 16 + j] = (bf16_t)v;
}

// ---------- layer2 epilog: h2 = relu(agg + b2) in place --------------------
__global__ void k_bias_relu2(float* __restrict__ agg, const float* __restrict__ b) {
    int tid = blockIdx.x * blockDim.x + threadIdx.x;
    if (tid >= N_NODES * OUT_CH) return;
    float v = agg[tid] + b[tid & 127];
    agg[tid] = v > 0.f ? v : 0.f;
}

// ---------- global mean pool accumulation ----------------------------------
__global__ void k_pool(const float* __restrict__ h2, const int* __restrict__ batch,
                       float* __restrict__ sums, float* __restrict__ cnt) {
    int tid = blockIdx.x * blockDim.x + threadIdx.x;
    if (tid >= N_NODES * OUT_CH) return;
    int n = tid >> 7, c = tid & 127;
    int g = batch[n];
    atomicAdd(&sums[(size_t)g * OUT_CH + c], h2[tid]);
    if (c == 0) atomicAdd(&cnt[g], 1.0f);
}

// ---------- batchnorm over graphs (single block, 128 channels) -------------
__global__ void k_bn(const float* __restrict__ sums, const float* __restrict__ cnt,
                     const float* __restrict__ gamma, const float* __restrict__ beta,
                     float* __restrict__ out) {
    int c = threadIdx.x;
    if (c >= OUT_CH) return;
    float s1 = 0.f, s2 = 0.f;
    for (int g = 0; g < N_GRAPHS; ++g) {
        float denom = cnt[g] > 1.f ? cnt[g] : 1.f;
        float p = sums[(size_t)g * OUT_CH + c] / denom;
        s1 += p; s2 += p * p;
    }
    float mu = s1 / N_GRAPHS;
    float var = s2 / N_GRAPHS - mu * mu;
    float inv = rsqrtf(var + BN_EPS);
    float gm = gamma[c], bt = beta[c];
    for (int g = 0; g < N_GRAPHS; ++g) {
        float denom = cnt[g] > 1.f ? cnt[g] : 1.f;
        float p = sums[(size_t)g * OUT_CH + c] / denom;
        out[(size_t)g * OUT_CH + c] = (p - mu) * inv * gm + bt;
    }
}

extern "C" void kernel_launch(void* const* d_in, const int* in_sizes, int n_in,
                              void* d_out, int out_size, void* d_ws, size_t ws_size,
                              hipStream_t stream) {
    const float* x        = (const float*)d_in[0];
    const int*   ei       = (const int*)  d_in[1];
    const int*   batch    = (const int*)  d_in[2];
    const float* W1       = (const float*)d_in[3];
    const float* att_src1 = (const float*)d_in[4];
    const float* att_dst1 = (const float*)d_in[5];
    const float* b1       = (const float*)d_in[6];
    const float* W2       = (const float*)d_in[7];
    const float* att_src2 = (const float*)d_in[8];
    const float* att_dst2 = (const float*)d_in[9];
    const float* b2       = (const float*)d_in[10];
    const float* gamma    = (const float*)d_in[11];
    const float* beta     = (const float*)d_in[12];
    float* out = (float*)d_out;

    // ---- workspace carve-up (256B aligned) ----
    char* ws = (char*)d_ws;
    size_t off = 0;
    auto take = [&](size_t bytes) -> void* {
        void* p = ws + off;
        off = (off + bytes + 255) & ~(size_t)255;
        return p;
    };
    bf16_t*   xp    = (bf16_t*)  take((size_t)MT * KB1 * 32 * 16 * sizeof(bf16_t));
    bf16_t*   w1p   = (bf16_t*)  take((size_t)16 * KB1 * 32 * 16 * sizeof(bf16_t));
    bf16_t*   w2p   = (bf16_t*)  take((size_t)8  * KB2 * 32 * 16 * sizeof(bf16_t));
    bf16_t*   h1p   = (bf16_t*)  take((size_t)MT * KB2 * 32 * 16 * sizeof(bf16_t));
    float*    hpre  = (float*)   take((size_t)N_NODES * HD1 * sizeof(float));
    float*    agg   = (float*)   take((size_t)N_NODES * HD1 * sizeof(float));
    float*    a_s   = (float*)   take((size_t)N_NODES * HEADS1 * sizeof(float));
    float*    a_d   = (float*)   take((size_t)N_NODES * HEADS1 * sizeof(float));
    unsigned* mkey  = (unsigned*)take((size_t)N_NODES * HEADS1 * sizeof(unsigned));
    float*    zsum  = (float*)   take((size_t)N_NODES * HEADS1 * sizeof(float));
    float*    sums  = (float*)   take((size_t)N_GRAPHS * OUT_CH * sizeof(float));
    float*    cnt   = (float*)   take((size_t)N_GRAPHS * sizeof(float));
    (void)ws_size; (void)n_in; (void)in_sizes; (void)out_size;

    const int TB = 256;

    // ---- pack operands to bf16 fragment layouts ----
    hipLaunchKernelGGL(k_pack_x, dim3((N_NODES * 64 + TB - 1) / TB), dim3(TB), 0, stream, x, xp);
    hipLaunchKernelGGL(k_pack_w, dim3((64 * HD1 + TB - 1) / TB), dim3(TB), 0, stream,
                       W1, w1p, IN_CH, 64, HD1, KB1);
    hipLaunchKernelGGL(k_pack_w, dim3((256 * OUT_CH + TB - 1) / TB), dim3(TB), 0, stream,
                       W2, w2p, HD1, 256, OUT_CH, KB2);

    // =================== layer 1 ===================
    hipMemsetAsync(agg,  0, (size_t)N_NODES * HD1 * sizeof(float), stream);
    hipMemsetAsync(mkey, 0, (size_t)N_NODES * HEADS1 * sizeof(unsigned), stream);
    hipMemsetAsync(zsum, 0, (size_t)N_NODES * HEADS1 * sizeof(float), stream);

    // hpre = x @ W1   (bf16 WMMA, f32 accumulate); 16 col tiles, 8 per block
    hipLaunchKernelGGL(k_gemm, dim3(MT, 2), dim3(128), 0, stream,
                       xp, w1p, hpre, KB1, HD1);
    hipLaunchKernelGGL(k_att, dim3((N_NODES * HEADS1 + TB - 1) / TB), dim3(TB), 0, stream,
                       hpre, att_src1, att_dst1, a_s, a_d, HEADS1, D1);

    int et1 = E_TOT * HEADS1;
    hipLaunchKernelGGL(k_edge_max,    dim3((et1 + TB - 1) / TB), dim3(TB), 0, stream,
                       ei, a_s, a_d, mkey, HEADS1);
    hipLaunchKernelGGL(k_edge_expsum, dim3((et1 + TB - 1) / TB), dim3(TB), 0, stream,
                       ei, a_s, a_d, mkey, zsum, HEADS1);
    hipLaunchKernelGGL(k_edge_agg, dim3((et1 * 32 + TB - 1) / TB), dim3(TB), 0, stream,
                       ei, a_s, a_d, mkey, zsum, hpre, agg, HEADS1, D1);

    hipLaunchKernelGGL(k_bias_relu_pack1, dim3((N_NODES * HD1 + TB - 1) / TB), dim3(TB),
                       0, stream, agg, b1, h1p);

    // =================== layer 2 ===================
    hipMemsetAsync(agg,  0, (size_t)N_NODES * OUT_CH * sizeof(float), stream);
    hipMemsetAsync(mkey, 0, (size_t)N_NODES * sizeof(unsigned), stream);
    hipMemsetAsync(zsum, 0, (size_t)N_NODES * sizeof(float), stream);

    // hpre(reused) = h1 @ W2 ; 8 col tiles, 8 per block
    hipLaunchKernelGGL(k_gemm, dim3(MT, 1), dim3(128), 0, stream,
                       h1p, w2p, hpre, KB2, OUT_CH);
    hipLaunchKernelGGL(k_att, dim3((N_NODES + TB - 1) / TB), dim3(TB), 0, stream,
                       hpre, att_src2, att_dst2, a_s, a_d, 1, OUT_CH);

    hipLaunchKernelGGL(k_edge_max,    dim3((E_TOT + TB - 1) / TB), dim3(TB), 0, stream,
                       ei, a_s, a_d, mkey, 1);
    hipLaunchKernelGGL(k_edge_expsum, dim3((E_TOT + TB - 1) / TB), dim3(TB), 0, stream,
                       ei, a_s, a_d, mkey, zsum, 1);
    hipLaunchKernelGGL(k_edge_agg, dim3(((size_t)E_TOT * 32 + TB - 1) / TB), dim3(TB),
                       0, stream, ei, a_s, a_d, mkey, zsum, hpre, agg, 1, OUT_CH);

    hipLaunchKernelGGL(k_bias_relu2, dim3((N_NODES * OUT_CH + TB - 1) / TB), dim3(TB),
                       0, stream, agg, b2);

    // =================== pool + BN ===================
    hipMemsetAsync(sums, 0, (size_t)N_GRAPHS * OUT_CH * sizeof(float), stream);
    hipMemsetAsync(cnt,  0, (size_t)N_GRAPHS * sizeof(float), stream);
    hipLaunchKernelGGL(k_pool, dim3((N_NODES * OUT_CH + TB - 1) / TB), dim3(TB), 0, stream,
                       agg, batch, sums, cnt);
    hipLaunchKernelGGL(k_bn, dim3(1), dim3(128), 0, stream, sums, cnt, gamma, beta, out);
}
